// RIN_63221918597498
// MI455X (gfx1250) — compile-verified, matrix operand order checked by reading, not run on hardware
//
#include <hip/hip_runtime.h>
#include <hip/hip_bf16.h>
#include <math.h>

// Problem dims
#define B_   512
#define T_   200
#define V_   99
#define H_   256
#define NBLK 32      // B_/16 : one block owns 16 batch rows for all T steps
#define NTH  512     // 16 wave32s per block

typedef __attribute__((ext_vector_type(16))) _Float16 v16h;
typedef __attribute__((ext_vector_type(8)))  float    v8f;

// ---------------- workspace layout (units: _Float16 elements) -------------
// one WMMA B-tile = 32 lanes * 16 halves = 512 halves, stored per-lane contiguous
#define F_WTD   0u          // KT=4,  NT=16 -> 32768
#define F_WHIST 32768u      // KT=8,  NT=7  -> 28672
#define F_WU    61440u      // KT=4,  NT=7  -> 14336
#define F_WV    75776u      // KT=4,  NT=7  -> 14336 (diag masked)
#define F_WR    90112u      // KT=4,  NT=7  -> 14336 (diag masked)
#define F_WIH   104448u     // KT=7,  NT=48 -> 172032
#define F_WHH   276480u     // KT=8,  NT=48 -> 196608
#define F_TOTAL 473088u     // ~946 KB of f16 frag weights

// ---------------- output layout (flat f32, reference return order) --------
#define BTV    10137600u    // 512*200*99
#define O_Y    10137600u
#define O_YS   10138112u
#define O_LOSS 10138624u
#define O_XUS  10138625u
#define O_XRS  20276225u

// ---------------- LDS ------------------------------------------------------
struct SMem {
  float    h[16][256];       // persistent GRU state (f32)
  float    gat[16][1024];    // r_sum | z_sum | gi_n | gh_n
  float    xt[16][112];
  float    mt[16][112];
  float    xhUnc[16][112];   // x_h, later unc_pre (disjoint lifetimes)
  float    xuP[16][112];
  float    xrP[16][112];
  _Float16 sA[16][256];      // d stage (k<128), later xcat stage (k<224)
  _Float16 sHg[16][256];     // decayed h, f16
  _Float16 sXbar[16][128];
  _Float16 sXr[16][128];
  _Float16 sU[16][128];
  float    red[32];          // per-wave partials: [0:16)=num, [16:32)=den
};                            // ~147 KB < 320 KB/WGP

// ---- A-fragment gather per ISA 7.12.2 (16-bit A 16x32 layout) -------------
// (compiler coalesces this into two ds_load_b128 per fragment)
__device__ __forceinline__ v16h load_a_frag(const _Float16* st, int stride,
                                            int kt, int lane){
  const int m = lane & 15, grp = lane >> 4;
  v16h a;
#pragma unroll
  for (int j = 0; j < 8; ++j){
    const int kb = kt*32 + ((j & 4) ? 16 : 0) + (grp ? 8 : 0) + (j & 3)*2;
    a[2*j]   = st[m*stride + kb];
    a[2*j+1] = st[m*stride + kb + 1];
  }
  return a;
}

// C[16 x 16] tile of  act[16 x K] @ W[N x K]^T  via f16 WMMA, f32 accum.
// B operand comes pre-swizzled per-lane from workspace (contiguous 32B/lane).
__device__ __forceinline__ v8f gemm_tile(const _Float16* st, int stride,
                                         const _Float16* __restrict__ fw,
                                         int KT, int NT, int nt, int lane){
  v8f c = {};
#pragma unroll 4
  for (int kt = 0; kt < KT; ++kt){
    v16h a = load_a_frag(st, stride, kt, lane);
    v16h b = *(const v16h*)(fw + ((size_t)(kt*NT + nt)*32 + lane)*16);
    c = __builtin_amdgcn_wmma_f32_16x16x32_f16(false, a, false, b,
                                               (short)0, c, false, false);
  }
  return c;
}

// ---- prep: pack W[N][K] (f32, optional zeroed diagonal) into per-lane
//      fragment order matching the B-operand register layout ---------------
__global__ void prep_frag(const float* __restrict__ W, _Float16* __restrict__ dst,
                          int N, int K, int NT, int KT, int maskDiag){
  int idx = blockIdx.x*blockDim.x + threadIdx.x;
  int total = KT*NT*512;
  if (idx >= total) return;
  int hi   = idx & 1;
  int j    = (idx >> 1) & 7;
  int lane = (idx >> 4) & 31;
  int tile = idx >> 9;
  int nt = tile % NT, kt = tile / NT;
  int grp = lane >> 4;
  int k = kt*32 + ((j & 4) ? 16 : 0) + (grp ? 8 : 0) + (j & 3)*2 + hi;
  int n = nt*16 + (lane & 15);
  float v = 0.f;
  if (k < K && n < N && !(maskDiag && n == k)) v = W[(size_t)n*K + k];
  dst[idx] = (_Float16)v;
}

struct RinParams {
  const float *x, *xhat, *u, *m, *d;
  const float *b_td, *b_hist, *b_v, *b_r, *b_u, *conv_w, *conv_b,
              *b_ih, *b_hh, *W_fc, *b_fc;
  const _Float16 *fwtd, *fwhist, *fwu, *fwv, *fwr, *fwih, *fwhh;
  float *out, *lossNum, *lossDen;
};

// ---- persistent main kernel: 32 blocks x 16 rows, all 200 steps ----------
__global__ __launch_bounds__(NTH) void rin_main(RinParams p){
  extern __shared__ char smem_raw[];
  SMem& S = *reinterpret_cast<SMem*>(smem_raw);
  const int tid  = threadIdx.x, lane = tid & 31, wave = tid >> 5;
  const int m16  = lane & 15,   grp  = lane >> 4;
  const int row0 = blockIdx.x * 16;

  const float cw0 = p.conv_w[0], cw1 = p.conv_w[1], cb0 = p.conv_b[0];

  for (int i = tid; i < 16*256; i += NTH) (&S.h[0][0])[i] = 0.f;
  __syncthreads();

  for (int t = 0; t < T_; ++t){
    // -- phase 0: stage x,m (f32) and d (f16); inputs are one-touch -> NT --
    for (int i = tid; i < 16*112; i += NTH){
      int r = i/112, v = i%112;
      float xv = 0.f, mv = 0.f;
      if (v < V_){
        size_t g = (size_t)(row0+r)*T_*V_ + (size_t)t*V_ + v;
        xv = __builtin_nontemporal_load(&p.x[g]);
        mv = __builtin_nontemporal_load(&p.m[g]);
      }
      S.xt[r][v] = xv; S.mt[r][v] = mv;
    }
    for (int i = tid; i < 16*128; i += NTH){
      int r = i/128, k = i%128;
      float dv = 0.f;
      if (k < V_) dv = __builtin_nontemporal_load(
                           &p.d[(size_t)(row0+r)*T_*V_ + (size_t)t*V_ + k]);
      S.sA[r][k] = (_Float16)dv;
    }
    // prefetch next step's input rows (global_prefetch_b8 hints into L2)
    if (t + 1 < T_ && tid < 160){
      int r = tid % 16, sel = (tid / 16) % 5, half = tid / 80;
      size_t g = (size_t)(row0+r)*T_*V_ + (size_t)(t+1)*V_ + half*64;
      const float* base = (sel==0) ? p.x : (sel==1) ? p.m : (sel==2) ? p.d
                        : (sel==3) ? p.xhat : p.u;
      __builtin_prefetch(&base[g], 0, 1);
    }
    __syncthreads();

    // -- phase 1: gamma = exp(-relu(d W_td^T + b_td)); h *= gamma ----------
    {
      int nt = wave;                       // 16 waves, 16 N-tiles of H=256
      v8f c = gemm_tile(&S.sA[0][0], 256, p.fwtd, 4, 16, nt, lane);
#pragma unroll
      for (int i = 0; i < 8; ++i){
        int r = i + 8*grp, n = nt*16 + m16;
        float gam = __expf(-fmaxf(c[i] + p.b_td[n], 0.f));
        float hv  = S.h[r][n] * gam;
        S.h[r][n]   = hv;
        S.sHg[r][n] = (_Float16)hv;
      }
    }
    __syncthreads();

    // -- phase 2: x_h = hg W_hist^T + b_hist  (7 N-tiles, wave-uniform) ----
    if (wave < 7){
      int nt = wave;
      v8f c = gemm_tile(&S.sHg[0][0], 256, p.fwhist, 8, 7, nt, lane);
#pragma unroll
      for (int i = 0; i < 8; ++i){
        int r = i + 8*grp, n = nt*16 + m16;
        if (n < V_) S.xhUnc[r][n] = c[i] + p.b_hist[n];
      }
    }
    __syncthreads();

    // -- phase 3: f16 stages x_r, xbar, u ---------------------------------
    for (int i = tid; i < 16*128; i += NTH){
      int r = i/128, v = i%128;
      float xr_ = 0.f, xb = 0.f, uu = 0.f;
      if (v < V_){
        size_t g = (size_t)(row0+r)*T_*V_ + (size_t)t*V_ + v;
        float xv = S.xt[r][v], mv = S.mt[r][v];
        xr_ = mv*xv + (1.f-mv)*S.xhUnc[r][v];
        xb  = mv*xv + (1.f-mv)*__builtin_nontemporal_load(&p.xhat[g]);
        uu  = __builtin_nontemporal_load(&p.u[g]);
      }
      S.sXr[r][v]   = (_Float16)xr_;
      S.sXbar[r][v] = (_Float16)xb;
      S.sU[r][v]    = (_Float16)uu;
    }
    __syncthreads();

    // -- phase 4: unc_pre / xu_pre / xr_pre  (3 x VxV GEMMs, 21 tile jobs) -
    for (int rep = 0; rep < 2; ++rep){
      int job = wave + 16*rep;
      if (job < 21){
        int which = job / 7, nt = job % 7;
        const _Float16* st = (which==0) ? &S.sU[0][0]
                           : (which==1) ? &S.sXbar[0][0] : &S.sXr[0][0];
        const _Float16* fwp = (which==0) ? p.fwu : (which==1) ? p.fwv : p.fwr;
        const float*   bias = (which==0) ? p.b_u : (which==1) ? p.b_v : p.b_r;
        float* dst = (which==0) ? &S.xhUnc[0][0]
                   : (which==1) ? &S.xuP[0][0] : &S.xrP[0][0];
        v8f c = gemm_tile(st, 128, fwp, 4, 7, nt, lane);
#pragma unroll
        for (int i = 0; i < 8; ++i){
          int r = i + 8*grp, n = nt*16 + m16;
          if (n < V_) dst[r*112 + n] = c[i] + bias[n];
        }
      }
    }
    __syncthreads();

    // -- phase 5: epilogue, NT output stores, xcat stage, loss partials ----
    float lnum = 0.f, lden = 0.f;
    for (int i = tid; i < 16*112; i += NTH){
      int r = i/112, v = i%112;
      if (v < V_){
        float xv  = S.xt[r][v], mv = S.mt[r][v];
        float unc = __expf(-fmaxf(S.xhUnc[r][v], 0.f));
        float xu  = S.xuP[r][v] * unc;
        float xrv = S.xrP[r][v];
        float xc  = cw0*xu + cw1*xrv + cb0;
        lnum += fabsf(xv - xc) * mv;
        lden += mv;
        float xi = mv*xv + (1.f-mv)*xc;
        size_t g = (size_t)(row0+r)*T_*V_ + (size_t)t*V_ + v;
        __builtin_nontemporal_store(xi,  &p.out[g]);
        __builtin_nontemporal_store(xu,  &p.out[O_XUS + g]);
        __builtin_nontemporal_store(xrv, &p.out[O_XRS + g]);
        S.sA[r][v]       = (_Float16)xi;   // xcat[:, 0:99)   = x_imp
        S.sA[r][V_ + v]  = (_Float16)mv;   // xcat[:, 99:198) = m
      }
    }
    for (int i = tid; i < 16*26; i += NTH){        // pad xcat cols 198..223
      int r = i/26, k = i%26;
      S.sA[r][198 + k] = (_Float16)0.f;
    }
    // wave-level reduction (wave32 shuffles), then one barrier
#pragma unroll
    for (int off = 16; off > 0; off >>= 1){
      lnum += __shfl_xor(lnum, off, 32);
      lden += __shfl_xor(lden, off, 32);
    }
    if (lane == 0){ S.red[wave] = lnum; S.red[16 + wave] = lden; }
    __syncthreads();   // covers both red[] and sA/xcat visibility for phase 6
    if (tid == 0){
      float n = 0.f, d2 = 0.f;
#pragma unroll
      for (int w = 0; w < 16; ++w){ n += S.red[w]; d2 += S.red[16 + w]; }
      p.lossNum[t*NBLK + blockIdx.x] = n;
      p.lossDen[t*NBLK + blockIdx.x] = d2;
    }

    // -- phase 6: GRU gates  gi = xcat W_ih^T,  gh = hg W_hh^T -------------
    for (int rep = 0; rep < 3; ++rep){
      int nt = wave + 16*rep;                      // 48 N-tiles of 3H=768
      v8f cg = gemm_tile(&S.sA[0][0],  256, p.fwih, 7, 48, nt, lane);
      v8f ch = gemm_tile(&S.sHg[0][0], 256, p.fwhh, 8, 48, nt, lane);
#pragma unroll
      for (int i = 0; i < 8; ++i){
        int r  = i + 8*grp;
        int n3 = nt*16 + m16;
        float gi = cg[i] + p.b_ih[n3];
        float gh = ch[i] + p.b_hh[n3];
        if (n3 < 512){                 // r,z gates only need the sum
          S.gat[r][n3] = gi + gh;
        } else {                       // n gate needs gi_n, gh_n separately
          S.gat[r][n3]       = gi;     // gi_n at cols [512,768)
          S.gat[r][n3 + 256] = gh;     // gh_n at cols [768,1024)
        }
      }
    }
    __syncthreads();

    // -- phase 7: GRU state update ----------------------------------------
    for (int i = tid; i < 16*256; i += NTH){
      int r = i/256, n = i%256;
      float rr = 1.f/(1.f + __expf(-S.gat[r][n]));
      float zz = 1.f/(1.f + __expf(-S.gat[r][256+n]));
      float nn = tanhf(S.gat[r][512+n] + rr*S.gat[r][768+n]);
      S.h[r][n] = (1.f - zz)*nn + zz*S.h[r][n];
    }
    __syncthreads();
  } // t loop

  // -- finale: y_out / y_score --------------------------------------------
  if (tid < 16){
    float acc = p.b_fc[0];
    for (int k = 0; k < 256; ++k) acc += S.h[tid][k] * p.W_fc[k];
    p.out[O_Y  + row0 + tid] = acc;
    p.out[O_YS + row0 + tid] = 1.f/(1.f + __expf(-acc));
  }
}

// ---- final loss reduction over (t, block) partials -----------------------
__global__ void rin_loss(const float* __restrict__ num,
                         const float* __restrict__ den,
                         float* __restrict__ out){
  __shared__ float red[256];
  float acc = 0.f;
  for (int t = (int)threadIdx.x; t < T_; t += 256){
    float n = 0.f, d = 0.f;
    for (int b = 0; b < NBLK; ++b){ n += num[t*NBLK + b]; d += den[t*NBLK + b]; }
    acc += n / (d + 1e-5f);
  }
  red[threadIdx.x] = acc; __syncthreads();
  for (int s = 128; s > 0; s >>= 1){
    if (threadIdx.x < s) red[threadIdx.x] += red[threadIdx.x+s];
    __syncthreads();
  }
  if (threadIdx.x == 0) out[O_LOSS] = red[0];
}

extern "C" void kernel_launch(void* const* d_in, const int* in_sizes, int n_in,
                              void* d_out, int out_size, void* d_ws, size_t ws_size,
                              hipStream_t stream){
  (void)in_sizes; (void)n_in; (void)out_size; (void)ws_size;
  const float* x      = (const float*)d_in[0];
  const float* xhat   = (const float*)d_in[1];
  const float* u      = (const float*)d_in[2];
  const float* m      = (const float*)d_in[3];
  const float* d      = (const float*)d_in[4];
  /* d_in[5] = y (unused) */
  const float* W_td   = (const float*)d_in[6];
  const float* b_td   = (const float*)d_in[7];
  const float* W_hist = (const float*)d_in[8];
  const float* b_hist = (const float*)d_in[9];
  const float* W_v    = (const float*)d_in[10];
  const float* b_v    = (const float*)d_in[11];
  const float* W_r    = (const float*)d_in[12];
  const float* b_r    = (const float*)d_in[13];
  const float* W_u    = (const float*)d_in[14];
  const float* b_u    = (const float*)d_in[15];
  const float* conv_w = (const float*)d_in[16];
  const float* conv_b = (const float*)d_in[17];
  const float* W_ih   = (const float*)d_in[18];
  const float* b_ih   = (const float*)d_in[19];
  const float* W_hh   = (const float*)d_in[20];
  const float* b_hh   = (const float*)d_in[21];
  const float* W_fc   = (const float*)d_in[22];
  const float* b_fc   = (const float*)d_in[23];

  _Float16* fw = (_Float16*)d_ws;
  float* lossNum = (float*)((char*)d_ws + (size_t)F_TOTAL*2);
  float* lossDen = lossNum + T_*NBLK;

  auto launch_prep = [&](const float* W, unsigned off, int N, int K,
                         int NT, int KT, int mask){
    int total = KT*NT*512;
    prep_frag<<<(total+255)/256, 256, 0, stream>>>(W, fw + off, N, K, NT, KT, mask);
  };
  launch_prep(W_td,   F_WTD,   H_,   V_,   16, 4, 0);
  launch_prep(W_hist, F_WHIST, V_,   H_,   7,  8, 0);
  launch_prep(W_u,    F_WU,    V_,   V_,   7,  4, 0);
  launch_prep(W_v,    F_WV,    V_,   V_,   7,  4, 1);   // off-diagonal mask
  launch_prep(W_r,    F_WR,    V_,   V_,   7,  4, 1);   // off-diagonal mask
  launch_prep(W_ih,   F_WIH,   3*H_, 2*V_, 48, 7, 0);
  launch_prep(W_hh,   F_WHH,   3*H_, H_,   48, 8, 0);

  RinParams p;
  p.x = x; p.xhat = xhat; p.u = u; p.m = m; p.d = d;
  p.b_td = b_td; p.b_hist = b_hist; p.b_v = b_v; p.b_r = b_r; p.b_u = b_u;
  p.conv_w = conv_w; p.conv_b = conv_b; p.b_ih = b_ih; p.b_hh = b_hh;
  p.W_fc = W_fc; p.b_fc = b_fc;
  p.fwtd = fw+F_WTD; p.fwhist = fw+F_WHIST; p.fwu = fw+F_WU;
  p.fwv = fw+F_WV; p.fwr = fw+F_WR; p.fwih = fw+F_WIH; p.fwhh = fw+F_WHH;
  p.out = (float*)d_out; p.lossNum = lossNum; p.lossDen = lossDen;

  static_assert(sizeof(SMem) <= 320*1024, "LDS budget");
  int smem = (int)sizeof(SMem);
  hipFuncSetAttribute(reinterpret_cast<const void*>(rin_main),
                      hipFuncAttributeMaxDynamicSharedMemorySize, smem);
  rin_main<<<dim3(NBLK), dim3(NTH), smem, stream>>>(p);
  rin_loss<<<dim3(1), dim3(256), 0, stream>>>(lossNum, lossDen, (float*)d_out);
}
